// MambaLayer_32452772889242
// MI455X (gfx1250) — compile-verified, matrix operand order checked by reading, not run on hardware
//
#include <hip/hip_runtime.h>
#include <hip/hip_bf16.h>

// ---------------------------------------------------------------------------
// MambaLayer on gfx1250 (MI455X): LN -> GEMM(w1) -> dwconv+SiLU -> GEMM(W_dt,
// softplus) -> B/C proj -> selective scan -> *SiLU(GEMM(v1)) -> GEMM(w2).
// Dense GEMMs run on v_wmma_f32_16x16x32_bf16 with double-buffered LDS tiles
// staged by GLOBAL_LOAD_ASYNC_TO_LDS_B128 (ASYNCcnt) when available.
// ---------------------------------------------------------------------------

typedef __attribute__((ext_vector_type(16))) __bf16 bf16x16;
typedef __attribute__((ext_vector_type(8)))  __bf16 bf16x8;
typedef __attribute__((ext_vector_type(8)))  float  f32x8;

#define BATCH 4
#define SEQL  2048
#define DIM   1024
#define NST   16
#define ROWS  (BATCH * SEQL)   // 8192

#if defined(__gfx1250__) && \
    __has_builtin(__builtin_amdgcn_global_load_async_to_lds_b128) && \
    __has_builtin(__builtin_amdgcn_s_wait_asynccnt)
#define ASYNC_OK 1
#else
#define ASYNC_OK 0
#endif

#if ASYNC_OK
typedef int v4i __attribute__((__vector_size__(16)));
typedef __attribute__((address_space(1))) v4i as1_v4i;
typedef __attribute__((address_space(3))) v4i as3_v4i;
__device__ __forceinline__ void async_ld_b128(const void* g, void* l) {
    __builtin_amdgcn_global_load_async_to_lds_b128(
        (as1_v4i*)g, (as3_v4i*)l, 0, 0);
}
__device__ __forceinline__ void wait_async() {
    __builtin_amdgcn_s_wait_asynccnt(0);
}
#else
__device__ __forceinline__ void wait_async() {}
#endif

__device__ __forceinline__ __bf16 f2bf(float f) {
    unsigned u = __builtin_bit_cast(unsigned, f);
    u += 0x7FFFu + ((u >> 16) & 1u);                 // round-to-nearest-even
    unsigned short h = (unsigned short)(u >> 16);
    return __builtin_bit_cast(__bf16, h);
}

__device__ __forceinline__ float silu_f(float x) {
    return x / (1.0f + __expf(-x));
}

__device__ __forceinline__ float activate(float x, int act) {
    if (act == 1) return silu_f(x);                              // SiLU
    if (act == 2) return (x > 20.0f) ? x : log1pf(__expf(x));    // softplus
    return x;
}

// ---------------------------------------------------------------------------
// Weight transpose + f32->bf16 convert: W[k][n] (KxN row-major) -> Wt[n][k].
// ---------------------------------------------------------------------------
__global__ void tconv_kernel(const float* __restrict__ W, __bf16* __restrict__ Wt) {
    __shared__ float tile[32][33];
    int n0 = blockIdx.x * 32;
    int k0 = blockIdx.y * 32;
    for (int r = threadIdx.y; r < 32; r += 8)
        tile[r][threadIdx.x] = W[(size_t)(k0 + r) * DIM + n0 + threadIdx.x];
    __syncthreads();
    for (int r = threadIdx.y; r < 32; r += 8)
        Wt[(size_t)(n0 + r) * DIM + k0 + threadIdx.x] = f2bf(tile[threadIdx.x][r]);
}

// ---------------------------------------------------------------------------
// LayerNorm over D=1024, one block per row, bf16 output.
// ---------------------------------------------------------------------------
__global__ void ln_kernel(const float* __restrict__ x,
                          const float* __restrict__ g,
                          const float* __restrict__ b,
                          __bf16* __restrict__ out) {
    int row = blockIdx.x;
    int tid = threadIdx.x;
    const float* xr = x + (size_t)row * DIM;
    float s = 0.0f, s2 = 0.0f;
    for (int i = tid; i < DIM; i += 256) {
        float v = xr[i];
        s += v; s2 += v * v;
    }
    __shared__ float rs[256], rs2[256];
    rs[tid] = s; rs2[tid] = s2;
    __syncthreads();
    for (int off = 128; off > 0; off >>= 1) {
        if (tid < off) { rs[tid] += rs[tid + off]; rs2[tid] += rs2[tid + off]; }
        __syncthreads();
    }
    float mu  = rs[0] * (1.0f / DIM);
    float var = rs2[0] * (1.0f / DIM) - mu * mu;
    float inv = rsqrtf(var + 1e-5f);
    __bf16* orow = out + (size_t)row * DIM;
    for (int i = tid; i < DIM; i += 256)
        orow[i] = f2bf((xr[i] - mu) * inv * g[i] + b[i]);
}

// ---------------------------------------------------------------------------
// bf16 WMMA GEMM: out[M x N] = act(A[M x K] @ Wt[N x K]^T + bias), f32 out.
// 128x128 tile / workgroup, 256 threads = 8 waves as 4(M) x 2(N); each wave
// owns 2x4 tiles of 16x16. K stepped by 32; next slab async-staged to the
// other LDS buffer while WMMAs consume the current one.
// ---------------------------------------------------------------------------
#define LDP 40   // LDS row pitch in bf16 (80B: 16B-aligned chunks)

__device__ __forceinline__ bf16x16 load_frag(const __bf16* base, int lane) {
    // 16-bit operand striping: lanes 0-15 rows 0-15 hold K {0..7,16..23},
    // lanes 16-31 same rows hold K {8..15,24..31}.
    int r  = lane & 15;
    int kh = (lane >> 4) * 8;
    const __bf16* p = base + r * LDP;
    bf16x8 lo = *(const bf16x8*)(p + kh);
    bf16x8 hi = *(const bf16x8*)(p + 16 + kh);
    bf16x16 v;
#pragma unroll
    for (int i = 0; i < 8; ++i) { v[i] = lo[i]; v[8 + i] = hi[i]; }
    return v;
}

__device__ __forceinline__ void stage_tile(const __bf16* ag, const __bf16* bg,
                                           __bf16 (*Asb)[LDP], __bf16 (*Bsb)[LDP],
                                           int srow, int scol) {
#if ASYNC_OK
    async_ld_b128(ag,     &Asb[srow][scol]);
    async_ld_b128(ag + 8, &Asb[srow][scol + 8]);
    async_ld_b128(bg,     &Bsb[srow][scol]);
    async_ld_b128(bg + 8, &Bsb[srow][scol + 8]);
#else
    int4 a0 = *(const int4*)(ag);
    int4 a1 = *(const int4*)(ag + 8);
    int4 b0 = *(const int4*)(bg);
    int4 b1 = *(const int4*)(bg + 8);
    *(int4*)&Asb[srow][scol]     = a0;
    *(int4*)&Asb[srow][scol + 8] = a1;
    *(int4*)&Bsb[srow][scol]     = b0;
    *(int4*)&Bsb[srow][scol + 8] = b1;
#endif
}

__global__ void __launch_bounds__(256)
gemm_bf16_kernel(const __bf16* __restrict__ A,   // M x K row-major
                 const __bf16* __restrict__ Wt,  // N x K row-major (pre-transposed)
                 const float*  __restrict__ bias,
                 float* __restrict__ out,
                 int M, int N, int K, int act) {
    __shared__ __bf16 As[2][128][LDP];
    __shared__ __bf16 Bs[2][128][LDP];

    const int m0 = blockIdx.y * 128;
    const int n0 = blockIdx.x * 128;
    const int tid    = threadIdx.x;
    const int lane   = tid & 31;
    const int waveId = tid >> 5;
    const int waveM  = waveId & 3;   // 4 waves over M, 32 rows each
    const int waveN  = waveId >> 2;  // 2 waves over N, 64 cols each

    // Staging map: thread t owns 16 bf16 (32B) of its (row, half) slot.
    const int srow = tid >> 1;
    const int scol = (tid & 1) * 16;
    const __bf16* agBase = A  + (size_t)(m0 + srow) * K + scol;
    const __bf16* bgBase = Wt + (size_t)(n0 + srow) * K + scol;

    f32x8 acc[2][4];
#pragma unroll
    for (int mi = 0; mi < 2; ++mi)
#pragma unroll
        for (int ni = 0; ni < 4; ++ni)
            acc[mi][ni] = (f32x8)0.0f;

    // Prologue: stage first K-slab into buffer 0.
    stage_tile(agBase, bgBase, As[0], Bs[0], srow, scol);
    wait_async();
    __syncthreads();

    for (int k0 = 0; k0 < K; k0 += 32) {
        const int cur = (k0 >> 5) & 1;

        // Overlap: async-stage next slab into the other buffer.
        if (k0 + 32 < K) {
            stage_tile(agBase + k0 + 32, bgBase + k0 + 32,
                       As[cur ^ 1], Bs[cur ^ 1], srow, scol);
            if (k0 + 64 < K) {   // global_prefetch_b8 two slabs ahead
                __builtin_prefetch(agBase + k0 + 64, 0, 1);
                __builtin_prefetch(bgBase + k0 + 64, 0, 1);
            }
        }

        bf16x16 af[2], bfr[4];
#pragma unroll
        for (int mi = 0; mi < 2; ++mi)
            af[mi] = load_frag(&As[cur][waveM * 32 + mi * 16][0], lane);
#pragma unroll
        for (int ni = 0; ni < 4; ++ni)
            bfr[ni] = load_frag(&Bs[cur][waveN * 64 + ni * 16][0], lane);

#pragma unroll
        for (int mi = 0; mi < 2; ++mi)
#pragma unroll
            for (int ni = 0; ni < 4; ++ni)
                acc[mi][ni] = __builtin_amdgcn_wmma_f32_16x16x32_bf16(
                    false, af[mi], false, bfr[ni],
                    (short)0, acc[mi][ni], false, false);

        wait_async();
        __syncthreads();
    }

    // Epilogue: C layout = 8 VGPRs, lane 0-15 -> M=v, lane 16-31 -> M=8+v.
    const int rsel = lane >> 4;
    const int cl   = lane & 15;
#pragma unroll
    for (int mi = 0; mi < 2; ++mi) {
#pragma unroll
        for (int ni = 0; ni < 4; ++ni) {
            int gm = m0 + waveM * 32 + mi * 16 + rsel * 8;
            int gn = n0 + waveN * 64 + ni * 16 + cl;
            float bs = bias[gn];
#pragma unroll
            for (int v = 0; v < 8; ++v)
                out[(size_t)(gm + v) * N + gn] = activate(acc[mi][ni][v] + bs, act);
        }
    }
}

// ---------------------------------------------------------------------------
// Depthwise conv (k=3, pad=1 along L) + SiLU; emits f32 + bf16.
// ---------------------------------------------------------------------------
__global__ void conv_kernel(const float* __restrict__ x1,
                            const float* __restrict__ cw,   // (D,1,3)
                            float* __restrict__ xc,
                            __bf16* __restrict__ xcb) {
    size_t idx = (size_t)blockIdx.x * 256 + threadIdx.x;   // B*L*D
    int d = (int)(idx & (DIM - 1));
    int l = (int)((idx >> 10) & (SEQL - 1));
    float c  = x1[idx];
    float lf = (l > 0)        ? x1[idx - DIM] : 0.0f;
    float rt = (l < SEQL - 1) ? x1[idx + DIM] : 0.0f;
    const float* w = cw + d * 3;
    float y = silu_f(lf * w[0] + c * w[1] + rt * w[2]);
    xc[idx]  = y;
    xcb[idx] = f2bf(y);
}

// ---------------------------------------------------------------------------
// B/C projection: bc[row][0:16] = xc @ W_B, bc[row][16:32] = xc @ W_C.
// ---------------------------------------------------------------------------
__global__ void bc_kernel(const float* __restrict__ xc,
                          const float* __restrict__ WB,
                          const float* __restrict__ WC,
                          float* __restrict__ bc) {
    int j   = threadIdx.x;                       // 0..31
    int row = blockIdx.x * 8 + threadIdx.y;      // 0..8191
    const float* W = (j < NST) ? WB : WC;
    int jj = j & (NST - 1);
    const float* xr = xc + (size_t)row * DIM;
    float acc = 0.0f;
#pragma unroll 4
    for (int k = 0; k < DIM; ++k)
        acc += xr[k] * W[k * NST + jj];
    bc[(size_t)row * 32 + j] = acc;
}

// ---------------------------------------------------------------------------
// Selective scan: one thread per (b,d) channel, 16 states in registers.
// B_t/C_t chunks (contiguous 2KB per 16 steps) async-staged to LDS.
// Output fused with * SiLU-gate and bf16-converted for the final GEMM.
// ---------------------------------------------------------------------------
__global__ void __launch_bounds__(256)
scan_kernel(const float* __restrict__ dt,
            const float* __restrict__ bc,
            const float* __restrict__ u,       // x_conv f32
            const float* __restrict__ A_log,
            const float* __restrict__ D_skip,
            const float* __restrict__ vgate,   // silu(x_norm @ v1 + b)
            __bf16* __restrict__ z) {
    const int b = blockIdx.x >> 2;
    const int d = (blockIdx.x & 3) * 256 + threadIdx.x;
    const int tid = threadIdx.x;

    float A[NST], h[NST];
#pragma unroll
    for (int n = 0; n < NST; ++n) {
        A[n] = -__expf(A_log[d * NST + n]);
        h[n] = 0.0f;
    }
    const float dsk = D_skip[d];

    __shared__ float sBC[16][32];   // [step][0:16]=B, [16:32]=C

    const size_t rowBase = (size_t)b * SEQL;
    for (int t0 = 0; t0 < SEQL; t0 += 16) {
        const float* chunk = bc + (rowBase + t0) * 32;     // contiguous 512 f32
#if ASYNC_OK
        if (tid < 128)
            async_ld_b128(chunk + tid * 4, &sBC[0][0] + tid * 4);
        wait_async();
#else
        for (int r = tid; r < 16 * 32; r += 256)
            sBC[r >> 5][r & 31] = chunk[r];
#endif
        __syncthreads();
#pragma unroll 1
        for (int i = 0; i < 16; ++i) {
            size_t off = (rowBase + t0 + i) * DIM + d;
            float dtv = dt[off];
            float uv  = u[off];
            float du  = dtv * uv;
            float y = 0.0f;
#pragma unroll
            for (int n = 0; n < NST; ++n) {
                float dA = __expf(dtv * A[n]);
                h[n] = dA * h[n] + du * sBC[i][n];
                y += h[n] * sBC[i][16 + n];
            }
            y += uv * dsk;
            z[off] = f2bf(y * vgate[off]);
        }
        __syncthreads();
    }
}

// ---------------------------------------------------------------------------
// Host-side launcher
// ---------------------------------------------------------------------------
extern "C" void kernel_launch(void* const* d_in, const int* in_sizes, int n_in,
                              void* d_out, int out_size, void* d_ws, size_t ws_size,
                              hipStream_t stream) {
    const float* x      = (const float*)d_in[0];
    const float* ln_g   = (const float*)d_in[1];
    const float* ln_b   = (const float*)d_in[2];
    const float* w1_w   = (const float*)d_in[3];
    const float* w1_b   = (const float*)d_in[4];
    const float* v1_w   = (const float*)d_in[5];
    const float* v1_b   = (const float*)d_in[6];
    const float* w2_w   = (const float*)d_in[7];
    const float* w2_b   = (const float*)d_in[8];
    const float* conv_w = (const float*)d_in[9];
    const float* A_log  = (const float*)d_in[10];
    const float* D_skip = (const float*)d_in[11];
    const float* W_dt   = (const float*)d_in[12];
    const float* b_dt   = (const float*)d_in[13];
    const float* W_B    = (const float*)d_in[14];
    const float* W_C    = (const float*)d_in[15];
    float* out = (float*)d_out;

    char* ws = (char*)d_ws;
    size_t off = 0;
    auto alloc = [&](size_t bytes) {
        size_t o = off;
        off = (off + bytes + 255) & ~(size_t)255;
        return (void*)(ws + o);
    };
    const size_t actB16 = (size_t)ROWS * DIM * sizeof(__bf16);   // 16 MB
    const size_t actF32 = (size_t)ROWS * DIM * sizeof(float);    // 32 MB
    const size_t wB16   = (size_t)DIM * DIM * sizeof(__bf16);    //  2 MB

    __bf16* xnorm_b = (__bf16*)alloc(actB16);
    __bf16* w1t     = (__bf16*)alloc(wB16);
    __bf16* v1t     = (__bf16*)alloc(wB16);
    __bf16* w2t     = (__bf16*)alloc(wB16);
    __bf16* wdtt    = (__bf16*)alloc(wB16);
    float*  x1      = (float*) alloc(actF32);
    float*  vgate   = (float*) alloc(actF32);
    float*  xc      = (float*) alloc(actF32);
    __bf16* xcb     = (__bf16*)alloc(actB16);
    float*  dtbuf   = (float*) alloc(actF32);
    float*  bcbuf   = (float*) alloc((size_t)ROWS * 32 * sizeof(float));
    __bf16* zb      = (__bf16*)alloc(actB16);
    (void)ws_size; (void)in_sizes; (void)n_in; (void)out_size;

    dim3 tGrid(DIM / 32, DIM / 32), tBlk(32, 8);
    tconv_kernel<<<tGrid, tBlk, 0, stream>>>(w1_w, w1t);
    tconv_kernel<<<tGrid, tBlk, 0, stream>>>(v1_w, v1t);
    tconv_kernel<<<tGrid, tBlk, 0, stream>>>(w2_w, w2t);
    tconv_kernel<<<tGrid, tBlk, 0, stream>>>(W_dt, wdtt);

    ln_kernel<<<ROWS, 256, 0, stream>>>(x, ln_g, ln_b, xnorm_b);

    dim3 gGrid(DIM / 128, ROWS / 128);  // (8, 64)
    gemm_bf16_kernel<<<gGrid, 256, 0, stream>>>(xnorm_b, w1t, w1_b, x1,
                                                ROWS, DIM, DIM, /*act=*/0);
    gemm_bf16_kernel<<<gGrid, 256, 0, stream>>>(xnorm_b, v1t, v1_b, vgate,
                                                ROWS, DIM, DIM, /*act=*/1);

    conv_kernel<<<(ROWS * DIM) / 256, 256, 0, stream>>>(x1, conv_w, xc, xcb);

    gemm_bf16_kernel<<<gGrid, 256, 0, stream>>>(xcb, wdtt, b_dt, dtbuf,
                                                ROWS, DIM, DIM, /*act=*/2);

    bc_kernel<<<ROWS / 8, dim3(32, 8), 0, stream>>>(xc, W_B, W_C, bcbuf);

    scan_kernel<<<BATCH * (DIM / 256), 256, 0, stream>>>(dtbuf, bcbuf, xc,
                                                         A_log, D_skip, vgate, zb);

    gemm_bf16_kernel<<<gGrid, 256, 0, stream>>>(zb, w2t, w2_b, out,
                                                ROWS, DIM, DIM, /*act=*/0);
}
